// MultiHeadAttention_65326452572164
// MI455X (gfx1250) — matrix operand order repaired and checked by reading, not compiled, not run on hardware
//
#include <hip/hip_runtime.h>
#include <hip/hip_bf16.h>

typedef __bf16 bf16;
typedef __attribute__((ext_vector_type(16))) __bf16 v16bf;
typedef __attribute__((ext_vector_type(8)))  float  v8f;

#define D_MODEL 768
#define NUM_HEADS 12
#define DK 64
#define SEQ 2048
#define BATCH 2

union FragBF { uint4 u[2]; v16bf v; };

// A-fragment (16x32, row-major source, stride in elements).
// lane m=lane&15 holds row (row0+m); K elems: [k0+16H .. k0+16H+15] contiguous
__device__ __forceinline__ v16bf load_a_frag(const bf16* base, int row0, int stride, int k0) {
  int lane = threadIdx.x & 31;
  int m = lane & 15, h = lane >> 4;
  const bf16* p = base + (size_t)(row0 + m) * stride + k0 + 16 * h;
  FragBF f;
  f.u[0] = *(const uint4*)p;
  f.u[1] = *(const uint4*)(p + 8);
  return f.v;
}

// B-fragment (32x16) loaded from B^T stored row-major [N x K].
// lane n=lane&15 holds column n; K elems: [k0+16H .. k0+16H+15] contiguous
__device__ __forceinline__ v16bf load_b_frag(const bf16* baseT, int n0, int stride, int k0) {
  int lane = threadIdx.x & 31;
  int n = lane & 15, h = lane >> 4;
  const bf16* p = baseT + (size_t)(n0 + n) * stride + k0 + 16 * h;
  FragBF f;
  f.u[0] = *(const uint4*)p;
  f.u[1] = *(const uint4*)(p + 8);
  return f.v;
}

__device__ __forceinline__ v8f wmma_bf16(v16bf a, v16bf b, v8f c) {
  return __builtin_amdgcn_wmma_f32_16x16x32_bf16(false, a, false, b, (short)0, c, false, false);
}

// ---------------- Kernel 0: fp32 -> bf16 conversions ----------------
__global__ void mha_convert_kernel(const float* __restrict__ x,
                                   const float* __restrict__ Wq,
                                   const float* __restrict__ Wk,
                                   const float* __restrict__ Wv,
                                   const float* __restrict__ Wo,
                                   bf16* __restrict__ xb,
                                   bf16* __restrict__ Wtb,
                                   bf16* __restrict__ Wob) {
  const int XN  = BATCH * SEQ * D_MODEL;        // 3145728
  const int WN  = NUM_HEADS * D_MODEL * DK;     // 589824 per matrix
  const int WON = D_MODEL * D_MODEL;            // 589824
  const int total = XN + 3 * WN + WON;
  for (int i = blockIdx.x * blockDim.x + threadIdx.x; i < total; i += gridDim.x * blockDim.x) {
    if (i < XN) { xb[i] = (bf16)x[i]; continue; }
    int j = i - XN;
    if (j < 3 * WN) {
      int mat = j / WN;
      int r = j % WN;
      int hh = r / (D_MODEL * DK);
      int r2 = r % (D_MODEL * DK);
      int d = r2 / DK;
      int k = r2 % DK;
      const float* W = (mat == 0) ? Wq : ((mat == 1) ? Wk : Wv);
      // store transposed: [mat, h, k(64), d(768)]
      Wtb[((size_t)(mat * NUM_HEADS + hh) * DK + k) * D_MODEL + d] = (bf16)W[r];
      continue;
    }
    int w = j - 3 * WN;
    Wob[w] = (bf16)Wo[w];
  }
}

// ---------------- Kernel 1: Q/K/V projections (WMMA, 32x64 tile/wave) ----------------
// Q,K stored [b,h,S,64]; V stored transposed [b,h,64,S].
__global__ void mha_qkv_kernel(const bf16* __restrict__ xb, const bf16* __restrict__ Wtb,
                               bf16* __restrict__ Qb, bf16* __restrict__ Kb,
                               bf16* __restrict__ Vtb) {
  int id = blockIdx.x;
  int st = id % (SEQ / 32);   id /= (SEQ / 32);
  int h  = id % NUM_HEADS;    id /= NUM_HEADS;
  int b  = id % BATCH;        id /= BATCH;
  int mat = id;               // 0=Q 1=K 2=V

  int s0 = st * 32;
  const bf16* A  = xb + (size_t)b * SEQ * D_MODEL;
  const bf16* Bt = Wtb + (size_t)(mat * NUM_HEADS + h) * DK * D_MODEL;

  v8f acc[2][4] = {};
  for (int k0 = 0; k0 < D_MODEL; k0 += 32) {
    v16bf a0 = load_a_frag(A, s0,      D_MODEL, k0);
    v16bf a1 = load_a_frag(A, s0 + 16, D_MODEL, k0);
#pragma unroll
    for (int nt = 0; nt < 4; ++nt) {
      v16bf bb = load_b_frag(Bt, nt * 16, D_MODEL, k0);
      acc[0][nt] = wmma_bf16(a0, bb, acc[0][nt]);
      acc[1][nt] = wmma_bf16(a1, bb, acc[1][nt]);
    }
  }

  int lane = threadIdx.x & 31;
  int n = lane & 15, half = lane >> 4;
  size_t hb = (size_t)(b * NUM_HEADS + h);
#pragma unroll
  for (int mi = 0; mi < 2; ++mi) {
#pragma unroll
    for (int nt = 0; nt < 4; ++nt) {
#pragma unroll
      for (int r = 0; r < 8; ++r) {
        int m = s0 + mi * 16 + r + 8 * half;
        int c = nt * 16 + n;
        bf16 val = (bf16)acc[mi][nt][r];
        if (mat == 0)      Qb[(hb * SEQ + m) * DK + c] = val;
        else if (mat == 1) Kb[(hb * SEQ + m) * DK + c] = val;
        else               Vtb[(hb * DK + c) * SEQ + m] = val;
      }
    }
  }
}

// ---------------- Kernel 2: causal flash attention (WMMA) ----------------
// One wave per (b,h,16-query block). Online softmax; P goes through LDS to
// convert C-layout -> A-fragment layout for the P@V WMMA.
__global__ void mha_attn_kernel(const bf16* __restrict__ Qb, const bf16* __restrict__ Kb,
                                const bf16* __restrict__ Vtb, bf16* __restrict__ Cc) {
  __shared__ __align__(16) bf16 Plds[16 * 32];

  int id = blockIdx.x;
  int qt = id % (SEQ / 16); id /= (SEQ / 16);
  int h  = id % NUM_HEADS;  id /= NUM_HEADS;
  int b  = id;
  int q0 = qt * 16;

  size_t hb = (size_t)(b * NUM_HEADS + h);
  const bf16* Q  = Qb  + hb * SEQ * DK;
  const bf16* K  = Kb  + hb * SEQ * DK;
  const bf16* Vt = Vtb + hb * DK * SEQ;

  int lane = threadIdx.x & 31;
  int n = lane & 15, half = lane >> 4;
  const float scale = 0.125f;  // 1/sqrt(64)

  v16bf qa0 = load_a_frag(Q, q0, DK, 0);
  v16bf qa1 = load_a_frag(Q, q0, DK, 32);

  v8f o0 = {}, o1 = {}, o2 = {}, o3 = {};
  float rm[8], rs[8];
#pragma unroll
  for (int r = 0; r < 8; ++r) { rm[r] = -1e30f; rs[r] = 0.0f; }

  for (int j0 = 0; j0 <= q0 + 15; j0 += 32) {
    v8f s0 = {}, s1 = {};
    { v16bf kb = load_b_frag(K, j0,      DK, 0);  s0 = wmma_bf16(qa0, kb, s0); }
    { v16bf kb = load_b_frag(K, j0,      DK, 32); s0 = wmma_bf16(qa1, kb, s0); }
    { v16bf kb = load_b_frag(K, j0 + 16, DK, 0);  s1 = wmma_bf16(qa0, kb, s1); }
    { v16bf kb = load_b_frag(K, j0 + 16, DK, 32); s1 = wmma_bf16(qa1, kb, s1); }

#pragma unroll
    for (int r = 0; r < 8; ++r) {
      int m = r + 8 * half;
      int gq = q0 + m;
      float v0 = s0[r] * scale; if (j0 + n > gq)      v0 = -1e30f;
      float v1 = s1[r] * scale; if (j0 + 16 + n > gq) v1 = -1e30f;
      float lmax = fmaxf(v0, v1);
#pragma unroll
      for (int msk = 1; msk <= 8; msk <<= 1)
        lmax = fmaxf(lmax, __shfl_xor(lmax, msk, 32));
      float nm = fmaxf(rm[r], lmax);
      float corr = __expf(rm[r] - nm);
      rm[r] = nm;
      float p0 = __expf(v0 - nm);
      float p1 = __expf(v1 - nm);
      float ls = p0 + p1;
#pragma unroll
      for (int msk = 1; msk <= 8; msk <<= 1)
        ls += __shfl_xor(ls, msk, 32);
      rs[r] = rs[r] * corr + ls;
      o0[r] *= corr; o1[r] *= corr; o2[r] *= corr; o3[r] *= corr;
      Plds[m * 32 + n]      = (bf16)p0;
      Plds[m * 32 + 16 + n] = (bf16)p1;
    }
    __syncthreads();
    v16bf pa = load_a_frag(Plds, 0, 32, 0);
    o0 = wmma_bf16(pa, load_b_frag(Vt, 0,  SEQ, j0), o0);
    o1 = wmma_bf16(pa, load_b_frag(Vt, 16, SEQ, j0), o1);
    o2 = wmma_bf16(pa, load_b_frag(Vt, 32, SEQ, j0), o2);
    o3 = wmma_bf16(pa, load_b_frag(Vt, 48, SEQ, j0), o3);
    __syncthreads();
  }

  // normalize and write concat layout [b, s, h*64 + f] as bf16
  bf16* out = Cc + (size_t)b * SEQ * D_MODEL + (size_t)h * DK;
#pragma unroll
  for (int r = 0; r < 8; ++r) {
    int m = r + 8 * half;
    float inv = 1.0f / rs[r];
    bf16* row = out + (size_t)(q0 + m) * D_MODEL;
    row[0 * 16 + n] = (bf16)(o0[r] * inv);
    row[1 * 16 + n] = (bf16)(o1[r] * inv);
    row[2 * 16 + n] = (bf16)(o2[r] * inv);
    row[3 * 16 + n] = (bf16)(o3[r] * inv);
  }
}

// ---------------- Kernel 3: output projection (WMMA, 32x64 tile/wave) + bias ----------------
__global__ void mha_proj_kernel(const bf16* __restrict__ Cc, const bf16* __restrict__ Wob,
                                const float* __restrict__ bo, float* __restrict__ out) {
  int id = blockIdx.x;
  int ng = id % (D_MODEL / 64); id /= (D_MODEL / 64);
  int mt = id;  // row tile over BATCH*SEQ = 4096 rows, 32 rows each
  int m0 = mt * 32, n0 = ng * 64;

  v8f acc[2][4] = {};
  for (int k0 = 0; k0 < D_MODEL; k0 += 32) {
    v16bf a0 = load_a_frag(Cc, m0,      D_MODEL, k0);
    v16bf a1 = load_a_frag(Cc, m0 + 16, D_MODEL, k0);
#pragma unroll
    for (int nt = 0; nt < 4; ++nt) {
      v16bf bb = load_b_frag(Wob, n0 + nt * 16, D_MODEL, k0);  // Wo row-major == B^T
      acc[0][nt] = wmma_bf16(a0, bb, acc[0][nt]);
      acc[1][nt] = wmma_bf16(a1, bb, acc[1][nt]);
    }
  }

  int lane = threadIdx.x & 31;
  int n = lane & 15, half = lane >> 4;
#pragma unroll
  for (int nt = 0; nt < 4; ++nt) {
    float bias = bo[n0 + nt * 16 + n];
#pragma unroll
    for (int mi = 0; mi < 2; ++mi) {
#pragma unroll
      for (int r = 0; r < 8; ++r) {
        int m = m0 + mi * 16 + r + 8 * half;
        out[(size_t)m * D_MODEL + n0 + nt * 16 + n] = acc[mi][nt][r] + bias;
      }
    }
  }
}

extern "C" void kernel_launch(void* const* d_in, const int* in_sizes, int n_in,
                              void* d_out, int out_size, void* d_ws, size_t ws_size,
                              hipStream_t stream) {
  const float* x  = (const float*)d_in[0];
  const float* Wq = (const float*)d_in[1];
  const float* Wk = (const float*)d_in[2];
  const float* Wv = (const float*)d_in[3];
  const float* Wo = (const float*)d_in[4];
  const float* bo = (const float*)d_in[5];
  float* out = (float*)d_out;

  char* ws = (char*)d_ws;
  size_t off = 0;
  bf16* xb  = (bf16*)(ws + off); off += (size_t)BATCH * SEQ * D_MODEL * 2;
  bf16* Wtb = (bf16*)(ws + off); off += (size_t)3 * NUM_HEADS * DK * D_MODEL * 2;
  bf16* Wob = (bf16*)(ws + off); off += (size_t)D_MODEL * D_MODEL * 2;
  bf16* Qb  = (bf16*)(ws + off); off += (size_t)BATCH * NUM_HEADS * SEQ * DK * 2;
  bf16* Kb  = (bf16*)(ws + off); off += (size_t)BATCH * NUM_HEADS * SEQ * DK * 2;
  bf16* Vtb = (bf16*)(ws + off); off += (size_t)BATCH * NUM_HEADS * DK * SEQ * 2;
  bf16* Cc  = (bf16*)(ws + off); off += (size_t)BATCH * SEQ * D_MODEL * 2;

  mha_convert_kernel<<<1024, 256, 0, stream>>>(x, Wq, Wk, Wv, Wo, xb, Wtb, Wob);
  mha_qkv_kernel<<<3 * BATCH * NUM_HEADS * (SEQ / 32), 32, 0, stream>>>(xb, Wtb, Qb, Kb, Vtb);
  mha_attn_kernel<<<BATCH * NUM_HEADS * (SEQ / 16), 32, 0, stream>>>(Qb, Kb, Vtb, Cc);
  mha_proj_kernel<<<(BATCH * SEQ / 32) * (D_MODEL / 64), 32, 0, stream>>>(Cc, Wob, bo, out);
}
